// GNNNodeClassifier_68478958568006
// MI455X (gfx1250) — compile-verified, matrix operand order checked by reading, not run on hardware
//
#include <hip/hip_runtime.h>
#include <hip/hip_bf16.h>

typedef __attribute__((ext_vector_type(16))) __bf16 v16bf;
typedef __attribute__((ext_vector_type(8)))  float  v8f;
typedef unsigned short u16;

#define NCLS 2
#define BN_EPS 1e-3f
#define L2_EPS 1e-12f

union Frag16 { v16bf v; unsigned u[8]; };

__device__ __forceinline__ u16 f2bf(float f) {
    unsigned u = __float_as_uint(f);
    u += 0x7FFFu + ((u >> 16) & 1u);          // round-to-nearest-even
    return (u16)(u >> 16);
}

// ---- BN (inference) -> bf16 cast --------------------------------------------
__global__ __launch_bounds__(256)
void bn_bf16_kernel(const float* __restrict__ x, const float* __restrict__ gamma,
                    const float* __restrict__ beta, const float* __restrict__ mean,
                    const float* __restrict__ var, u16* __restrict__ out,
                    int M, int D) {
    int i = blockIdx.x * 256 + threadIdx.x;
    if (i >= M * D) return;
    int d = i % D;
    float v = (x[i] - mean[d]) * rsqrtf(var[d] + BN_EPS) * gamma[d] + beta[d];
    out[i] = f2bf(v);
}

// BN over concat([xa, xb]) with 256-wide stats -> bf16 [M,256]
__global__ __launch_bounds__(256)
void bn2_bf16_kernel(const float* __restrict__ xa, const float* __restrict__ xb,
                     const float* __restrict__ gamma, const float* __restrict__ beta,
                     const float* __restrict__ mean, const float* __restrict__ var,
                     u16* __restrict__ out, int M) {
    int i = blockIdx.x * 256 + threadIdx.x;
    if (i >= M * 256) return;
    int d = i % 256, r = i / 256;
    float xv = (d < 128) ? xa[(size_t)r * 128 + d] : xb[(size_t)r * 128 + (d - 128)];
    float v = (xv - mean[d]) * rsqrtf(var[d] + BN_EPS) * gamma[d] + beta[d];
    out[i] = f2bf(v);
}

// ---- weight fp32 -> bf16 ----------------------------------------------------
__global__ __launch_bounds__(256)
void cvt_bf16_kernel(const float* __restrict__ w, u16* __restrict__ out, int n) {
    int i = blockIdx.x * 256 + threadIdx.x;
    if (i < n) out[i] = f2bf(w[i]);
}

__global__ __launch_bounds__(256)
void fill0_kernel(float* __restrict__ p, int n) {
    int i = blockIdx.x * 256 + threadIdx.x;
    if (i < n) p[i] = 0.0f;
}

// ---- GEMM: out[M,128] = gelu(xb[M,K] @ W[K,128] + bias), bf16 WMMA ----------
// One wave computes a 16x128 strip. W staged in LDS column-major [128][K].
__device__ __forceinline__ int kmap(int v, int half) {
    // 16-bit A/B fragment K index for VGPR v, pair base (ISA 7.12.2 layout)
    return ((v >> 2) << 4) + (half << 3) + ((v & 3) << 1);
}

__global__ __launch_bounds__(256)
void gemm_bias_gelu_kernel(const u16* __restrict__ xb, const u16* __restrict__ Wb,
                           const float* __restrict__ bias, float* __restrict__ out,
                           int M, int K) {
    extern __shared__ u16 Wlds[];                 // [128][K], col-major per output col
    int tid = threadIdx.x;
    int total = K * 128;
    for (int i = tid; i < total; i += 256) {      // coalesced load, transpose into LDS
        int k = i >> 7, n = i & 127;              // Wb is [K,128] row-major
        Wlds[n * K + k] = Wb[i];
    }
    __syncthreads();

    int wave = tid >> 5, lane = tid & 31;
    int half = lane >> 4, l16 = lane & 15;
    int rowBase = (blockIdx.x * 8 + wave) * 16;

    v8f acc[8];
#pragma unroll
    for (int n = 0; n < 8; ++n)
#pragma unroll
        for (int r = 0; r < 8; ++r) acc[n][r] = 0.0f;

    int ar = rowBase + l16;                       // A fragment row (lanes hold M=0..15)
    if (ar >= M) ar = M - 1;                      // clamp; EXEC must stay all-ones
    const u16* xrow = xb + (size_t)ar * K;

    int steps = K >> 5;
    for (int ks = 0; ks < steps; ++ks) {
        Frag16 a;
#pragma unroll
        for (int v = 0; v < 8; ++v)
            a.u[v] = *(const unsigned*)(xrow + ks * 32 + kmap(v, half));
#pragma unroll
        for (int n = 0; n < 8; ++n) {
            Frag16 b;
            const u16* wcol = Wlds + (size_t)(n * 16 + l16) * K;
#pragma unroll
            for (int v = 0; v < 8; ++v)
                b.u[v] = *(const unsigned*)(wcol + ks * 32 + kmap(v, half));
            acc[n] = __builtin_amdgcn_wmma_f32_16x16x32_bf16(
                false, a.v, false, b.v, (short)0, acc[n], false, false);
        }
    }

#pragma unroll
    for (int n = 0; n < 8; ++n) {
        int col = n * 16 + l16;
        float bv = bias[col];
#pragma unroll
        for (int r = 0; r < 8; ++r) {
            int row = rowBase + r + half * 8;     // C layout: VGPR r -> M=r / r+8
            if (row < M) {
                float v = acc[n][r] + bv;
                float g = 0.5f * v * (1.0f + erff(v * 0.70710678118654752f));
                out[(size_t)row * 128 + col] = g;
            }
        }
    }
}

// ---- edge aggregation: agg[tgt] += msgs[src] * w, wave per edge -------------
__global__ __launch_bounds__(256)
void aggregate_kernel(const float* __restrict__ msgs, const int* __restrict__ edges,
                      const float* __restrict__ ew, float* __restrict__ agg, int E) {
    int e = blockIdx.x * 8 + (threadIdx.x >> 5);
    if (e >= E) return;
    int lane = threadIdx.x & 31;
    int tgt = edges[e];
    int src = edges[E + e];
    float w = ew[e];
    const float* m = msgs + (size_t)src * 128;
    float* a = agg + (size_t)tgt * 128;
#pragma unroll
    for (int j = 0; j < 4; ++j) {
        int d = lane + 32 * j;
        atomicAdd(a + d, m[d] * w);
    }
}

// ---- x = h / ||h||_2 + x  (row l2-normalize + residual), wave per row -------
__global__ __launch_bounds__(256)
void l2res_kernel(const float* __restrict__ h, float* __restrict__ x, int M) {
    int row = blockIdx.x * 8 + (threadIdx.x >> 5);
    if (row >= M) return;
    int lane = threadIdx.x & 31;
    const float* hr = h + (size_t)row * 128;
    float* xr = x + (size_t)row * 128;
    float v[4], ss = 0.0f;
#pragma unroll
    for (int j = 0; j < 4; ++j) { v[j] = hr[lane + 32 * j]; ss += v[j] * v[j]; }
#pragma unroll
    for (int off = 16; off > 0; off >>= 1) ss += __shfl_xor(ss, off, 32);
    float inv = rsqrtf(fmaxf(ss, L2_EPS));
#pragma unroll
    for (int j = 0; j < 4; ++j) xr[lane + 32 * j] += v[j] * inv;
}

// ---- logits: out[B,2] = x[idx] @ W[128,2] + b -------------------------------
__global__ __launch_bounds__(256)
void logits_kernel(const float* __restrict__ x, const int* __restrict__ idx,
                   const float* __restrict__ W, const float* __restrict__ b,
                   float* __restrict__ out, int B) {
    int i = blockIdx.x * 256 + threadIdx.x;
    if (i >= B * NCLS) return;
    int r = i / NCLS, c = i % NCLS;
    const float* xr = x + (size_t)idx[r] * 128;
    float s = b[c];
#pragma unroll 4
    for (int k = 0; k < 128; ++k) s += xr[k] * W[k * NCLS + c];
    out[i] = s;
}

extern "C" void kernel_launch(void* const* d_in, const int* in_sizes, int n_in,
                              void* d_out, int out_size, void* d_ws, size_t ws_size,
                              hipStream_t stream) {
    (void)n_in; (void)out_size; (void)ws_size;
    const int N = in_sizes[0] / 128;
    const int E = in_sizes[1] / 2;
    const int B = in_sizes[3];

    const float* nodef = (const float*)d_in[0];
    const int*   edges = (const int*)d_in[1];
    const float* ew    = (const float*)d_in[2];
    const int*   nidx  = (const int*)d_in[3];
    float* outp        = (float*)d_out;

    // params flat order: 6 FFNs (pre, c1.prep, c1.upd, c2.prep, c2.upd, post)
    // x 2 layers x {gamma,beta,mean,var,W,b}; then logits {W,b}
    auto pp = [&](int f, int l, int k) -> const float* {
        return (const float*)d_in[4 + f * 12 + l * 6 + k];
    };
    const float* logW = (const float*)d_in[76];
    const float* logB = (const float*)d_in[77];
    auto dinOf = [](int f, int l) { return ((f == 2 || f == 4) && l == 0) ? 256 : 128; };

    // workspace carve
    size_t off = 0;
    auto carve = [&](size_t bytes) -> char* {
        char* p = (char*)d_ws + off;
        off += (bytes + 255) & ~(size_t)255;
        return p;
    };
    u16*   xb  = (u16*)carve((size_t)N * 256 * sizeof(u16));
    float* h0  = (float*)carve((size_t)N * 128 * sizeof(float));
    float* t0  = (float*)carve((size_t)N * 128 * sizeof(float));
    float* t1  = (float*)carve((size_t)N * 128 * sizeof(float));
    float* agg = (float*)carve((size_t)N * 128 * sizeof(float));
    u16* wbf[6][2];
    for (int f = 0; f < 6; ++f)
        for (int l = 0; l < 2; ++l)
            wbf[f][l] = (u16*)carve((size_t)dinOf(f, l) * 128 * sizeof(u16));

    // convert all GEMM weights fp32 -> bf16
    for (int f = 0; f < 6; ++f)
        for (int l = 0; l < 2; ++l) {
            int cnt = dinOf(f, l) * 128;
            cvt_bf16_kernel<<<(cnt + 255) / 256, 256, 0, stream>>>(pp(f, l, 4), wbf[f][l], cnt);
        }

    // one FFN layer: BN->bf16 cast, then WMMA GEMM with fused bias+GELU
    auto run_layer = [&](const float* xin, const float* xcat, int f, int l, float* dst) {
        int K = dinOf(f, l);
        int tot = N * K;
        if (xcat)
            bn2_bf16_kernel<<<(tot + 255) / 256, 256, 0, stream>>>(
                xin, xcat, pp(f, l, 0), pp(f, l, 1), pp(f, l, 2), pp(f, l, 3), xb, N);
        else
            bn_bf16_kernel<<<(tot + 255) / 256, 256, 0, stream>>>(
                xin, pp(f, l, 0), pp(f, l, 1), pp(f, l, 2), pp(f, l, 3), xb, N, K);
        int grid = (N + 127) / 128;                    // 8 waves x 16 rows per block
        size_t lds = (size_t)K * 128 * sizeof(u16);    // 32KB / 64KB dynamic LDS
        gemm_bias_gelu_kernel<<<grid, 256, lds, stream>>>(xb, wbf[f][l], pp(f, l, 5), dst, N, K);
    };

    // pre FFN: node_features -> h0
    run_layer(nodef, nullptr, 0, 0, t0);
    run_layer(t0, nullptr, 0, 1, h0);

    // graph conv (prepare per-NODE, gather via edges, update on concat, l2+residual)
    auto conv = [&](int fp, int fu) {
        run_layer(h0, nullptr, fp, 0, t0);
        run_layer(t0, nullptr, fp, 1, t1);             // t1 = ffn_prepare(x) per node
        int tot = N * 128;
        fill0_kernel<<<(tot + 255) / 256, 256, 0, stream>>>(agg, tot);
        aggregate_kernel<<<(E + 7) / 8, 256, 0, stream>>>(t1, edges, ew, agg, E);
        run_layer(h0, agg, fu, 0, t0);                 // concat(x, agg) -> 256-wide
        run_layer(t0, nullptr, fu, 1, t1);             // t1 = h
        l2res_kernel<<<(N + 7) / 8, 256, 0, stream>>>(t1, h0, N);  // h0 = h/||h|| + h0
    };
    conv(1, 2);
    conv(3, 4);

    // post FFN: h0 -> t1
    run_layer(h0, nullptr, 5, 0, t0);
    run_layer(t0, nullptr, 5, 1, t1);

    // logits
    logits_kernel<<<(B * NCLS + 255) / 256, 256, 0, stream>>>(t1, nidx, logW, logB, outp, B);
}
